// SparseMoE_18176301597238
// MI455X (gfx1250) — matrix operand order repaired and checked by reading, not compiled, not run on hardware
//
#include <hip/hip_runtime.h>
#include <hip/hip_bf16.h>
#include <math.h>

// ---------------------------------------------------------------------------
// SparseMoE for MI455X (gfx1250): router + fused expert MLP with bf16 WMMA.
// Weights pre-converted to bf16 in workspace (one-shot) when ws allows;
// fallback converts on the fly. Dense-over-experts (matches reference);
// combine weight folded into h. MT=32 token tile: each weight B-fragment
// feeds two WMMAs (two M halves) -> halves L2 weight traffic per FLOP.
// ---------------------------------------------------------------------------

#define NTOK   8192   // BATCH * SEQ
#define HID    1024
#define INTER  2048
#define NEXP   8

typedef __attribute__((ext_vector_type(16))) __bf16 v16bf;
typedef __attribute__((ext_vector_type(8)))  float  v8f;

// ---------------------------------------------------------------------------
// Kernel 1: router. One wave (32 lanes) per token.
// ---------------------------------------------------------------------------
__global__ __launch_bounds__(256) void router_kernel(
    const float* __restrict__ x, const float* __restrict__ w_gate,
    float* __restrict__ logits_out, float* __restrict__ combine) {
  __shared__ float wg[NEXP * HID];  // 32 KB
  int tid = threadIdx.x;
  for (int i = tid; i < NEXP * HID; i += 256) wg[i] = w_gate[i];
  __syncthreads();

  int wave = tid >> 5, lane = tid & 31;
  int t = blockIdx.x * 8 + wave;
  const float* xr = x + (size_t)t * HID;

  float acc[NEXP];
#pragma unroll
  for (int e = 0; e < NEXP; ++e) acc[e] = 0.f;
  for (int h = lane; h < HID; h += 32) {
    float xv = xr[h];
#pragma unroll
    for (int e = 0; e < NEXP; ++e) acc[e] = fmaf(xv, wg[e * HID + h], acc[e]);
  }
#pragma unroll
  for (int e = 0; e < NEXP; ++e) {
#pragma unroll
    for (int off = 16; off >= 1; off >>= 1)
      acc[e] += __shfl_xor(acc[e], off, 32);
  }

  if (lane == 0) {
    float m = acc[0];
#pragma unroll
    for (int e = 1; e < NEXP; ++e) m = fmaxf(m, acc[e]);
    float p[NEXP], s = 0.f;
#pragma unroll
    for (int e = 0; e < NEXP; ++e) { p[e] = __expf(acc[e] - m); s += p[e]; }
    float inv = 1.f / s;
#pragma unroll
    for (int e = 0; e < NEXP; ++e) p[e] *= inv;
    int i0 = 0;
#pragma unroll
    for (int e = 1; e < NEXP; ++e) if (p[e] > p[i0]) i0 = e;
    int i1 = (i0 == 0) ? 1 : 0;
#pragma unroll
    for (int e = 0; e < NEXP; ++e) if (e != i0 && p[e] > p[i1]) i1 = e;
    float wsum = p[i0] + p[i1];
    float* cmb = combine + (size_t)t * NEXP;
#pragma unroll
    for (int e = 0; e < NEXP; ++e) cmb[e] = 0.f;
    cmb[i0] = p[i0] / wsum;
    cmb[i1] = p[i1] / wsum;
    float* lo = logits_out + (size_t)t * NEXP;
#pragma unroll
    for (int e = 0; e < NEXP; ++e) lo[e] = acc[e];
  }
}

// ---------------------------------------------------------------------------
// Kernel 2: one-shot f32 -> bf16 weight conversion (bandwidth bound, ~9us).
// ---------------------------------------------------------------------------
__global__ __launch_bounds__(256) void cvt_kernel(
    const float* __restrict__ src, __bf16* __restrict__ dst, int n4) {
  int i = blockIdx.x * 256 + threadIdx.x;
  if (i >= n4) return;
  float4 v = ((const float4*)src)[i];
  union { __bf16 h[4]; uint2 u; } o;
  o.h[0] = (__bf16)v.x; o.h[1] = (__bf16)v.y;
  o.h[2] = (__bf16)v.z; o.h[3] = (__bf16)v.w;
  ((uint2*)dst)[i] = o.u;
}

// ---------------------------------------------------------------------------
// WMMA fragment helpers (16x16x32 bf16, wave32; A and B both use contiguous k = 16*(lane/16) + s)
// ---------------------------------------------------------------------------

// A fragment (16x32 bf16) from a row-major bf16 tile in LDS (rows r0..r0+15).
// Lane L(0-15): M=L, K {0..15}; lanes 16-31: M=L-16, K {16..31}.
// Per-lane data = two 16B-contiguous runs -> two ds_load_b128.
__device__ __forceinline__ v16bf load_a_lds(const __bf16* base, int ldm,
                                            int kb, int lane) {
  int r = lane & 15;
  const __bf16* row = base + r * ldm + kb + ((lane & 16) ? 16 : 0);
  union { v16bf v; uint4 q[2]; } f;
  f.q[0] = *(const uint4*)(row);        // K 0..7  (8 bf16)
  f.q[1] = *(const uint4*)(row + 8);    // K 8..15 (8 bf16)
  return f.v;
}

// B fragment (32x16) from row-major bf16 weights: B(k,n) = W[nbase+n][kb+k].
// Per-lane data = 16 contiguous bf16 -> two global_load_b128.
__device__ __forceinline__ v16bf load_b_frag(const __bf16* base, int stride,
                                             int kb, int lane) {
  int n = lane & 15;
  const __bf16* p = base + (size_t)n * stride + kb + ((lane & 16) ? 16 : 0);
  union { v16bf v; uint4 q[2]; } f;
  f.q[0] = *(const uint4*)(p);
  f.q[1] = *(const uint4*)(p + 8);
  return f.v;
}

// Fallback: B fragment from row-major f32 weights, cvt in registers.
__device__ __forceinline__ v16bf load_b_frag(const float* base, int stride,
                                             int kb, int lane) {
  int n = lane & 15;
  const float* p = base + (size_t)n * stride + kb + ((lane & 16) ? 16 : 0);
  union { v16bf v; __bf16 h[16]; } f;
#pragma unroll
  for (int j = 0; j < 8; ++j) {
    float2 w = *(const float2*)(p + 2 * j);
    f.h[2 * j]     = (__bf16)w.x;
    f.h[2 * j + 1] = (__bf16)w.y;
  }
  return f.v;
}

// ---------------------------------------------------------------------------
// Kernel 3: fused MoE MLP. Block = 256 threads (8 waves), 32-token tile.
// Output tile [32,1024] f32 lives in registers across all 8 experts.
// ---------------------------------------------------------------------------
#define MT     32
#define ICHUNK 128
#define LDX    (HID + 8)
#define LDH    (ICHUNK + 8)

template <typename WT>
__global__ __launch_bounds__(256, 1) void moe_kernel(
    const float* __restrict__ x, const WT* __restrict__ w_fc,
    const WT* __restrict__ w_proj, const float* __restrict__ combine,
    float* __restrict__ out) {
  __shared__ __bf16 xs[MT * LDX];   // 32 x 1032 bf16 ~ 66 KB
  __shared__ __bf16 hs[MT * LDH];   // 32 x 136 bf16  ~ 8.7 KB
  __shared__ float  comb[MT];

  int tid = threadIdx.x;
  int wave = tid >> 5, lane = tid & 31;
  int token0 = blockIdx.x * MT;

  // Stage x tile as bf16 in LDS.
  for (int i = tid; i < MT * HID; i += 256) {
    int r = i >> 10, c = i & (HID - 1);
    xs[r * LDX + c] = (__bf16)x[(size_t)(token0 + r) * HID + c];
  }

  // This wave owns out columns [wave*128, wave*128+128):
  // acc[2t]   = rows 0..15,  col tile t
  // acc[2t+1] = rows 16..31, col tile t
  v8f acc[16];
#pragma unroll
  for (int t = 0; t < 16; ++t) acc[t] = (v8f){};

  for (int e = 0; e < NEXP; ++e) {
    __syncthreads();  // also covers initial xs fill on e==0
    if (tid < MT) comb[tid] = combine[(size_t)(token0 + tid) * NEXP + e];
    __syncthreads();

    const WT* wfc_e = w_fc + (size_t)e * INTER * HID;
    const WT* wpj_e = w_proj + (size_t)e * HID * INTER;

    for (int ic = 0; ic < INTER; ic += ICHUNK) {
      // --- Phase A: h[0:32, ic+wave*16 : +16] = x_tile @ Wfc^T (K=1024) ---
      // One B fragment feeds both M halves.
      v8f c0 = {}, c1 = {};
      const WT* bbase = wfc_e + (size_t)(ic + wave * 16) * HID;
      for (int kb = 0; kb < HID; kb += 32) {
        v16bf b  = load_b_frag(bbase, HID, kb, lane);
        v16bf a0 = load_a_lds(xs, LDX, kb, lane);
        v16bf a1 = load_a_lds(xs + 16 * LDX, LDX, kb, lane);
        c0 = __builtin_amdgcn_wmma_f32_16x16x32_bf16(false, a0, false, b,
                                                     (short)0, c0, false, false);
        c1 = __builtin_amdgcn_wmma_f32_16x16x32_bf16(false, a1, false, b,
                                                     (short)0, c1, false, false);
      }

      __syncthreads();  // previous chunk's hs fully consumed
      {
        int n = lane & 15, hi = (lane >> 4) * 8;
#pragma unroll
        for (int i = 0; i < 8; ++i) {
          int row = hi + i;  // C layout: VGPR i -> M=i (+8 for upper lanes)
          float v0 = c0[i];
          float g0 = 0.5f * v0 * (1.0f + erff(v0 * 0.70710678118654752f));
          hs[row * LDH + wave * 16 + n] = (__bf16)(g0 * comb[row]);
          int row1 = row + 16;
          float v1 = c1[i];
          float g1 = 0.5f * v1 * (1.0f + erff(v1 * 0.70710678118654752f));
          hs[row1 * LDH + wave * 16 + n] = (__bf16)(g1 * comb[row1]);
        }
      }
      __syncthreads();

      // --- Phase B: out_acc += h_chunk @ Wproj[:, ic:ic+128]^T (K=128) ---
      // kb outer / t inner: two A fragments feed 16 WMMAs, B reused x2.
#pragma unroll
      for (int kb = 0; kb < ICHUNK; kb += 32) {
        v16bf a0 = load_a_lds(hs, LDH, kb, lane);
        v16bf a1 = load_a_lds(hs + 16 * LDH, LDH, kb, lane);
#pragma unroll
        for (int t = 0; t < 8; ++t) {
          int cb = wave * 128 + t * 16;
          const WT* pbase = wpj_e + (size_t)cb * INTER + ic;
          v16bf b = load_b_frag(pbase, INTER, kb, lane);
          acc[2 * t] = __builtin_amdgcn_wmma_f32_16x16x32_bf16(
              false, a0, false, b, (short)0, acc[2 * t], false, false);
          acc[2 * t + 1] = __builtin_amdgcn_wmma_f32_16x16x32_bf16(
              false, a1, false, b, (short)0, acc[2 * t + 1], false, false);
        }
      }
    }
  }

  // Write the finished [32,1024] f32 output tile (sum over all experts).
  {
    int n = lane & 15, hi = (lane >> 4) * 8;
#pragma unroll
    for (int t = 0; t < 8; ++t) {
      int cb = wave * 128 + t * 16;
#pragma unroll
      for (int i = 0; i < 8; ++i) {
        int row = hi + i;
        out[(size_t)(token0 + row) * HID + cb + n] = acc[2 * t][i];
        out[(size_t)(token0 + row + 16) * HID + cb + n] = acc[2 * t + 1][i];
      }
    }
  }
}

// ---------------------------------------------------------------------------
extern "C" void kernel_launch(void* const* d_in, const int* in_sizes, int n_in,
                              void* d_out, int out_size, void* d_ws,
                              size_t ws_size, hipStream_t stream) {
  const float* x      = (const float*)d_in[0];  // [2,4096,1024]
  const float* w_gate = (const float*)d_in[1];  // [8,1024]
  const float* w_fc   = (const float*)d_in[2];  // [8,2048,1024]
  const float* w_proj = (const float*)d_in[3];  // [8,1024,2048]

  float* out    = (float*)d_out;                 // [8192,1024]
  float* logits = out + (size_t)NTOK * HID;      // [8192,8]

  const size_t nW       = (size_t)NEXP * INTER * HID;           // 16.78M each
  const size_t cmbBytes = (size_t)NTOK * NEXP * sizeof(float);  // 256 KB
  const size_t need     = cmbBytes + 2 * nW * sizeof(__bf16);   // ~67.4 MB

  float* combine = (float*)d_ws;

  router_kernel<<<NTOK / 8, 256, 0, stream>>>(x, w_gate, logits, combine);

  if (ws_size >= need) {
    // Fast path: one-shot bf16 weight copies, then bf16-weight GEMM kernel.
    __bf16* wfc16 = (__bf16*)((char*)d_ws + cmbBytes);
    __bf16* wpj16 = wfc16 + nW;
    int n4 = (int)(nW / 4);
    int grid = (n4 + 255) / 256;
    cvt_kernel<<<grid, 256, 0, stream>>>(w_fc, wfc16, n4);
    cvt_kernel<<<grid, 256, 0, stream>>>(w_proj, wpj16, n4);
    moe_kernel<__bf16><<<NTOK / MT, 256, 0, stream>>>(x, wfc16, wpj16,
                                                      combine, out);
  } else {
    // Fallback: stream f32 weights, convert in registers.
    moe_kernel<float><<<NTOK / MT, 256, 0, stream>>>(x, w_fc, w_proj,
                                                     combine, out);
  }
}
